// RelMultiHeadAttn_4286377361569
// MI455X (gfx1250) — matrix operand order repaired and checked by reading, not compiled, not run on hardware
//
#include <hip/hip_runtime.h>

// ---------------- problem constants ----------------
#define S   2048   // QLEN == RLEN == KLEN
#define BSZ 2
#define NH  16
#define DH  64
#define DM  1024

// ---------------- workspace layout (bytes) ----------------
#define OFF_Q1 (size_t)0                    // [BSZ][NH][S][DH] bf16 : q + r_w_bias
#define OFF_Q2 (size_t)(8u  << 20)          // [BSZ][NH][S][DH] bf16 : q + r_r_bias
#define OFF_K  (size_t)(16u << 20)          // [BSZ][NH][S][DH] bf16
#define OFF_VT (size_t)(24u << 20)          // [BSZ][NH][DH][S] bf16 (transposed V)
#define OFF_RK (size_t)(32u << 20)          // [NH][S][DH] bf16
#define OFF_AV (size_t)(36u << 20)          // [S*BSZ][DM] bf16 attn_vec
#define OFF_Y  (size_t)(44u << 20)          // [S*BSZ][DM] fp32 attn_out (pre-LN)
#define WS_NEEDED (size_t)(60u << 20)

typedef __attribute__((ext_vector_type(16))) __bf16 v16bf;
typedef __attribute__((ext_vector_type(8)))  float  v8f;
typedef __attribute__((ext_vector_type(4)))  unsigned int v4u;
typedef __attribute__((ext_vector_type(8)))  int          v8i;
typedef __attribute__((ext_vector_type(4)))  int          v4i;

struct __align__(16) U128 { unsigned int x, y, z, w; };

union FragB {
  v16bf bf;
  U128  q[2];
};

__device__ __forceinline__ unsigned short f2bf(float f) {
  unsigned int u = __float_as_uint(f);
  u += 0x7FFFu + ((u >> 16) & 1u);          // round-to-nearest-even
  return (unsigned short)(u >> 16);
}

__device__ __forceinline__ v8f wmma_bf16(const FragB& a, const FragB& b, v8f c) {
  return __builtin_amdgcn_wmma_f32_16x16x32_bf16(false, a.bf, false, b.bf,
                                                 (short)0, c, false, false);
}

__device__ __forceinline__ v8f zero8() {
  v8f z;
#pragma unroll
  for (int i = 0; i < 8; ++i) z[i] = 0.f;
  return z;
}

// ---------------------------------------------------------------------
// Tensor Data Mover: issue a 2D bf16 tile load (tile_y rows x tile_x
// elems, row stride `stride_elems`) from global into LDS at lds_addr.
// D# layout per CDNA5 ISA 8.3/8.4 (group0: count/lds/global/type,
// group1: data_size=2B, tensor dims, tile dims, 48-bit dim0 stride).
// Tracked by TENSORcnt; pair with __builtin_amdgcn_s_wait_tensorcnt.
// ---------------------------------------------------------------------
__device__ __forceinline__ void tdm_load_2d(unsigned lds_addr, const void* gptr,
                                            unsigned tile_x, unsigned tile_y,
                                            unsigned stride_elems)
{
  unsigned long long ga = (unsigned long long)(size_t)gptr;
  v4u g0;
  g0[0] = 1u;                                             // count=1, user mode
  g0[1] = lds_addr;                                       // LDS byte address
  g0[2] = (unsigned)(ga & 0xFFFFFFFFu);                   // global_addr[31:0]
  g0[3] = (unsigned)((ga >> 32) & 0x01FFFFFFu) | (2u << 30); // addr[56:32] | type=2

  unsigned td0 = stride_elems;                            // tensor dim0 (row len)
  unsigned td1 = tile_y;                                  // tensor dim1 (rows)
  v8i g1;
  g1[0] = (int)(1u << 16);                                // data_size=1 -> 2 bytes
  g1[1] = (int)((td0 & 0xFFFFu) << 16);                   // tensor_dim0[15:0]
  g1[2] = (int)(((td0 >> 16) & 0xFFFFu) | ((td1 & 0xFFFFu) << 16));
  g1[3] = (int)(((td1 >> 16) & 0xFFFFu) | ((tile_x & 0xFFFFu) << 16)); // tile_dim0
  g1[4] = (int)(tile_y & 0xFFFFu);                        // tile_dim1 (tile_dim2=0)
  g1[5] = (int)stride_elems;                              // dim0_stride[31:0]
  g1[6] = 0;                                              // dim0_stride[47:32], dim1_stride lo
  g1[7] = 0;

  v4i gz4; gz4[0] = 0; gz4[1] = 0; gz4[2] = 0; gz4[3] = 0;
#if __clang_major__ >= 23
  v8i gz8;
#pragma unroll
  for (int i = 0; i < 8; ++i) gz8[i] = 0;
  __builtin_amdgcn_tensor_load_to_lds(g0, g1, gz4, gz4, gz8, 0);
#else
  __builtin_amdgcn_tensor_load_to_lds(g0, g1, gz4, gz4, 0);
#endif
}

// =====================================================================
// Kernel 1/2: projection GEMM  C = A(fp32) @ W(fp32), bf16 WMMA, fp32 acc
//   mode 0: A=h [4096x1024], W=W_qkv [1024x3072] -> scatter q1/q2/k/vT
//   mode 1: A=r [2048x1024], W=W_r   [1024x1024] -> scatter rk
// Block 256 (8 waves), tile 128(M) x 64(N), BK=32.
// =====================================================================
__global__ __launch_bounds__(256) void proj_gemm(
    const float* __restrict__ A, const float* __restrict__ W,
    int K, int N, int mode,
    const float* __restrict__ rwb, const float* __restrict__ rrb,
    unsigned short* __restrict__ q1, unsigned short* __restrict__ q2,
    unsigned short* __restrict__ kbuf, unsigned short* __restrict__ vt,
    unsigned short* __restrict__ rkbuf)
{
  __shared__ __align__(16) unsigned short sA[128 * 40];  // [m][k] bf16, pad 40
  __shared__ __align__(16) unsigned short sB[64 * 40];   // [n][k] bf16 (transposed)

  const int t = threadIdx.x;
  const int lane = t & 31, w = t >> 5;
  const int L15 = lane & 15, hh = lane >> 4;
  const int M0 = blockIdx.x * 128, N0 = blockIdx.y * 64;

  v8f acc[4];
#pragma unroll
  for (int s = 0; s < 4; ++s) acc[s] = zero8();

  const int m = w * 16 + L15;   // this lane's A-fragment row within the tile

  for (int k0 = 0; k0 < K; k0 += 32) {
#pragma unroll
    for (int rpt = 0; rpt < 16; ++rpt) {            // stage A 128x32
      int idx = t + 256 * rpt;
      int row = idx >> 5, kk = idx & 31;
      sA[row * 40 + kk] = f2bf(A[(size_t)(M0 + row) * K + (k0 + kk)]);
    }
#pragma unroll
    for (int rpt = 0; rpt < 8; ++rpt) {             // stage W 32x64, transposed
      int idx = t + 256 * rpt;
      int kk = idx >> 6, nn = idx & 63;
      sB[nn * 40 + kk] = f2bf(W[(size_t)(k0 + kk) * N + (N0 + nn)]);
    }
    __syncthreads();

    FragB fa;                                        // A frag: V0-7=K[16h..16h+16)
    fa.q[0] = *(const U128*)&sA[m * 40 + 16 * hh];
    fa.q[1] = *(const U128*)&sA[m * 40 + 16 * hh + 8];
#pragma unroll
    for (int s = 0; s < 4; ++s) {
      FragB fb;                                      // B frag: V0-7=K[16h..16h+16)
      int c = s * 16 + L15;
      fb.q[0] = *(const U128*)&sB[c * 40 + 16 * hh];
      fb.q[1] = *(const U128*)&sB[c * 40 + 16 * hh + 8];
      acc[s] = wmma_bf16(fa, fb, acc[s]);
    }
    __syncthreads();
  }

  // Epilogue scatter (C layout: row = v + 8*(lane/16), col = lane%16)
#pragma unroll
  for (int s = 0; s < 4; ++s) {
#pragma unroll
    for (int v = 0; v < 8; ++v) {
      int row = M0 + w * 16 + v + 8 * hh;
      int col = N0 + s * 16 + L15;
      float cval = acc[s][v];
      if (mode == 0) {
        int i = row >> 1, b = row & 1;               // row = i*BSZ + b
        int which = col >> 10, rem = col & 1023;     // 0:q 1:k 2:v
        int nh = rem >> 6, d = rem & 63;
        size_t hb = ((size_t)(b * NH + nh) * S + i) * DH + d;
        if (which == 0) {
          q1[hb] = f2bf(cval + rwb[nh * DH + d]);
          q2[hb] = f2bf(cval + rrb[nh * DH + d]);
        } else if (which == 1) {
          kbuf[hb] = f2bf(cval);
        } else {
          vt[((size_t)(b * NH + nh) * DH + d) * S + i] = f2bf(cval);
        }
      } else {
        int nh = col >> 6, d = col & 63;
        rkbuf[((size_t)nh * S + row) * DH + d] = f2bf(cval);
      }
    }
  }
}

// =====================================================================
// Kernel 3: fused flash attention with exact Transformer-XL rel-shift.
// Grid (S/64, NH, BSZ), 128 threads = 4 waves, 16 q-rows each.
// K and V^T chunks are DMA'd into double-buffered LDS by the Tensor Data
// Mover (wave 0 issues, s_wait_tensorcnt + barrier gates); AC/BD/PV all
// run on v_wmma_f32_16x16x32_bf16. BD uses two 16x48 P panels (rows i,
// rows i+1) -> LDS -> per-element rel-shift gather.
// =====================================================================
__global__ __launch_bounds__(128) void attn_kernel(
    const unsigned short* __restrict__ q1, const unsigned short* __restrict__ q2,
    const unsigned short* __restrict__ kbuf, const unsigned short* __restrict__ vt,
    const unsigned short* __restrict__ rkbuf, unsigned short* __restrict__ av)
{
  __shared__ __align__(16) float          sPa[4][16 * 49];
  __shared__ __align__(16) float          sPb[4][16 * 49];
  __shared__ __align__(16) unsigned short sProb[4][16 * 32];
  __shared__ __align__(16) unsigned short sK[2][32 * DH];  // K chunk [j][d]
  __shared__ __align__(16) unsigned short sV[2][DH * 32];  // V^T chunk [d][j]

  const int t = threadIdx.x;
  const int lane = t & 31, w = t >> 5;
  const int L15 = lane & 15, hh = lane >> 4;
  const int ib = blockIdx.x, n = blockIdx.y, b = blockIdx.z;
  const int bn = b * NH + n;
  const int i0 = ib * 64 + w * 16;

  const unsigned short* qb1 = q1    + (size_t)bn * S * DH;
  const unsigned short* qb2 = q2    + (size_t)bn * S * DH;
  const unsigned short* kb  = kbuf  + (size_t)bn * S * DH;
  const unsigned short* vb  = vt    + (size_t)bn * DH * S;
  const unsigned short* rkb = rkbuf + (size_t)n  * S * DH;

  // ---- load Q fragments once: rows i0.. (biased q1,q2) and rows i0+1.. (q2) ----
  FragB a1[2], a2[2], a2p[2];
  {
    int mr  = i0 + L15;
    int mr1 = min(mr + 1, S - 1);                    // row i0+16 never used at i0=S-16
#pragma unroll
    for (int ks = 0; ks < 2; ++ks) {
      size_t o  = (size_t)mr  * DH + ks * 32 + 16 * hh;
      size_t o1 = (size_t)mr1 * DH + ks * 32 + 16 * hh;
      a1[ks].q[0]  = *(const U128*)(qb1 + o);  a1[ks].q[1]  = *(const U128*)(qb1 + o + 8);
      a2[ks].q[0]  = *(const U128*)(qb2 + o);  a2[ks].q[1]  = *(const U128*)(qb2 + o + 8);
      a2p[ks].q[0] = *(const U128*)(qb2 + o1); a2p[ks].q[1] = *(const U128*)(qb2 + o1 + 8);
    }
  }

  float rmax[8], lsum[8];
#pragma unroll
  for (int v = 0; v < 8; ++v) { rmax[v] = -1e30f; lsum[v] = 0.f; }
  v8f oacc[4];
#pragma unroll
  for (int d4 = 0; d4 < 4; ++d4) oacc[d4] = zero8();

  // TDM double-buffered pipeline: issue chunks 0 and 1 up front (wave 0 only;
  // EXEC is ignored by tensor ops, TENSORcnt is per-wave).
  if (w == 0) {
    tdm_load_2d((unsigned)(size_t)&sK[0][0], kb,           DH, 32, DH);
    tdm_load_2d((unsigned)(size_t)&sV[0][0], vb,           32, DH, S);
    tdm_load_2d((unsigned)(size_t)&sK[1][0], kb + 32 * DH, DH, 32, DH);
    tdm_load_2d((unsigned)(size_t)&sV[1][0], vb + 32,      32, DH, S);
  }

  const int NCH = S / 32;
  for (int c = 0; c < NCH; ++c) {
    const int j0  = c * 32;
    const int buf = c & 1;

    // chunk c complete (chunk c+1's two tensor ops may still be in flight)
    if (w == 0) __builtin_amdgcn_s_wait_tensorcnt(2);
    __syncthreads();

    // ---------------- AC = (q + r_w_bias) . K^T ----------------
    v8f ac[2];
#pragma unroll
    for (int s = 0; s < 2; ++s) {
      ac[s] = zero8();
#pragma unroll
      for (int ks = 0; ks < 2; ++ks) {
        FragB fb;
        int jr = s * 16 + L15;
        fb.q[0] = *(const U128*)&sK[buf][jr * DH + ks * 32 + 16 * hh];
        fb.q[1] = *(const U128*)&sK[buf][jr * DH + ks * 32 + 16 * hh + 8];
        ac[s] = wmma_bf16(a1[ks], fb, ac[s]);
      }
    }

    // ---------------- BD panels: P[r,m] = (q_r + r_r_bias) . rk_m ----------------
    // rel-shift: j<=i -> P[i, S-1-i+j]; j==i+1 -> 0; j>i+1 -> P[i+1, j-i-2]
    const bool hasPa = (j0 <= i0 + 15);
    const int  pa0   = S - 16 - i0 + j0;
    if (hasPa) {
      __builtin_prefetch(rkb + (size_t)min(pa0, S - 1) * DH, 0, 1);
#pragma unroll
      for (int ct = 0; ct < 3; ++ct) {
        v8f p = zero8();
#pragma unroll
        for (int ks = 0; ks < 2; ++ks) {
          FragB fb;
          int mcol = min(pa0 + ct * 16 + L15, S - 1);
          size_t o = (size_t)mcol * DH + ks * 32 + 16 * hh;
          fb.q[0] = *(const U128*)(rkb + o);
          fb.q[1] = *(const U128*)(rkb + o + 8);
          p = wmma_bf16(a2[ks], fb, p);
        }
#pragma unroll
        for (int v = 0; v < 8; ++v)
          sPa[w][(v + 8 * hh) * 49 + ct * 16 + L15] = p[v];
      }
    }
    const bool hasPb = (j0 + 31 >= i0 + 2);
    const int  pb0   = max(0, j0 - i0 - 17);
    if (hasPb) {
#pragma unroll
      for (int ct = 0; ct < 3; ++ct) {
        v8f p = zero8();
#pragma unroll
        for (int ks = 0; ks < 2; ++ks) {
          FragB fb;
          int mcol = min(pb0 + ct * 16 + L15, S - 1);
          size_t o = (size_t)mcol * DH + ks * 32 + 16 * hh;
          fb.q[0] = *(const U128*)(rkb + o);
          fb.q[1] = *(const U128*)(rkb + o + 8);
          p = wmma_bf16(a2p[ks], fb, p);
        }
#pragma unroll
        for (int v = 0; v < 8; ++v)
          sPb[w][(v + 8 * hh) * 49 + ct * 16 + L15] = p[v];
      }
    }

    // ---------------- score assembly + streaming softmax ----------------
    float sc[2][8];
#pragma unroll
    for (int s = 0; s < 2; ++s) {
#pragma unroll
      for (int v = 0; v < 8; ++v) {
        int r = v + 8 * hh;
        int i = i0 + r;
        int j = j0 + s * 16 + L15;
        float bd;
        if (j <= i)          bd = sPa[w][r * 49 + (15 - r + s * 16 + L15)];
        else if (j == i + 1) bd = 0.f;
        else                 bd = sPb[w][r * 49 + (j - i - 2 - pb0)];
        sc[s][v] = (ac[s][v] + bd) * 0.125f;         // 1/sqrt(64)
      }
    }
    float alph[8];
#pragma unroll
    for (int v = 0; v < 8; ++v) {
      float mx = fmaxf(sc[0][v], sc[1][v]);
#pragma unroll
      for (int msk = 1; msk <= 8; msk <<= 1)
        mx = fmaxf(mx, __shfl_xor(mx, msk, 32));     // reduce across 16-lane half
      float mnew = fmaxf(rmax[v], mx);
      alph[v] = __expf(rmax[v] - mnew);
      rmax[v] = mnew;
      float ps = 0.f;
      sc[0][v] = __expf(sc[0][v] - mnew); ps += sc[0][v];
      sc[1][v] = __expf(sc[1][v] - mnew); ps += sc[1][v];
#pragma unroll
      for (int msk = 1; msk <= 8; msk <<= 1)
        ps += __shfl_xor(ps, msk, 32);
      lsum[v] = lsum[v] * alph[v] + ps;
    }
#pragma unroll
    for (int d4 = 0; d4 < 4; ++d4)
#pragma unroll
      for (int v = 0; v < 8; ++v)
        oacc[d4][v] *= alph[v];

    // ---- prob (acc layout) -> A-fragment via wave-private LDS (DS in-order) ----
#pragma unroll
    for (int s = 0; s < 2; ++s)
#pragma unroll
      for (int v = 0; v < 8; ++v)
        sProb[w][(v + 8 * hh) * 32 + s * 16 + L15] = f2bf(sc[s][v]);
    FragB fp;
    fp.q[0] = *(const U128*)&sProb[w][L15 * 32 + 16 * hh];
    fp.q[1] = *(const U128*)&sProb[w][L15 * 32 + 16 * hh + 8];

    // ---------------- O += prob . V  (V^T chunk staged in LDS) ----------------
#pragma unroll
    for (int d4 = 0; d4 < 4; ++d4) {
      FragB fb;
      int d = d4 * 16 + L15;
      fb.q[0] = *(const U128*)&sV[buf][d * 32 + 16 * hh];
      fb.q[1] = *(const U128*)&sV[buf][d * 32 + 16 * hh + 8];
      oacc[d4] = wmma_bf16(fp, fb, oacc[d4]);
    }

    // all waves done with buf -> wave 0 may overwrite it with chunk c+2
    __syncthreads();
    if (w == 0 && c + 2 < NCH) {
      tdm_load_2d((unsigned)(size_t)&sK[buf][0], kb + (size_t)(c + 2) * 32 * DH,
                  DH, 32, DH);
      tdm_load_2d((unsigned)(size_t)&sV[buf][0], vb + (size_t)(c + 2) * 32,
                  32, DH, S);
    }
  }

  // ---------------- write attn_vec [i*BSZ+b][n*64+d] (bf16) ----------------
  float invl[8];
#pragma unroll
  for (int v = 0; v < 8; ++v) invl[v] = 1.f / lsum[v];
#pragma unroll
  for (int d4 = 0; d4 < 4; ++d4) {
#pragma unroll
    for (int v = 0; v < 8; ++v) {
      int i = i0 + v + 8 * hh;
      int d = d4 * 16 + L15;
      av[(size_t)(i * BSZ + b) * DM + n * DH + d] = f2bf(oacc[d4][v] * invl[v]);
    }
  }
}

// =====================================================================
// Kernel 4: attn_out = attn_vec(bf16) @ W_o(fp32), fp32 out.
// =====================================================================
__global__ __launch_bounds__(256) void out_gemm(
    const unsigned short* __restrict__ Abf, const float* __restrict__ W,
    float* __restrict__ Y)
{
  __shared__ __align__(16) unsigned short sB[64 * 40];
  const int t = threadIdx.x;
  const int lane = t & 31, w = t >> 5;
  const int L15 = lane & 15, hh = lane >> 4;
  const int M0 = blockIdx.x * 128, N0 = blockIdx.y * 64;

  v8f acc[4];
#pragma unroll
  for (int s = 0; s < 4; ++s) acc[s] = zero8();

  const int mr = M0 + w * 16 + L15;

  for (int k0 = 0; k0 < DM; k0 += 32) {
#pragma unroll
    for (int rpt = 0; rpt < 8; ++rpt) {
      int idx = t + 256 * rpt;
      int kk = idx >> 6, nn = idx & 63;
      sB[nn * 40 + kk] = f2bf(W[(size_t)(k0 + kk) * DM + (N0 + nn)]);
    }
    __syncthreads();
    FragB fa;
    size_t o = (size_t)mr * DM + k0 + 16 * hh;
    fa.q[0] = *(const U128*)(Abf + o);
    fa.q[1] = *(const U128*)(Abf + o + 8);
#pragma unroll
    for (int s = 0; s < 4; ++s) {
      FragB fb;
      int c = s * 16 + L15;
      fb.q[0] = *(const U128*)&sB[c * 40 + 16 * hh];
      fb.q[1] = *(const U128*)&sB[c * 40 + 16 * hh + 8];
      acc[s] = wmma_bf16(fa, fb, acc[s]);
    }
    __syncthreads();
  }
#pragma unroll
  for (int s = 0; s < 4; ++s)
#pragma unroll
    for (int v = 0; v < 8; ++v) {
      int row = M0 + w * 16 + v + 8 * hh;
      int col = N0 + s * 16 + L15;
      Y[(size_t)row * DM + col] = acc[s][v];
    }
}

// =====================================================================
// Kernel 5: out = LayerNorm(h + attn_out), one block per row of 1024.
// =====================================================================
__global__ __launch_bounds__(256) void ln_kernel(
    const float* __restrict__ h, const float* __restrict__ y,
    const float* __restrict__ gamma, const float* __restrict__ beta,
    float* __restrict__ out)
{
  __shared__ float rs[256], rq[256];
  const int row = blockIdx.x, t = threadIdx.x;
  float x[4], s = 0.f, sq = 0.f;
#pragma unroll
  for (int e = 0; e < 4; ++e) {
    int c = t + e * 256;
    x[e] = h[(size_t)row * DM + c] + y[(size_t)row * DM + c];
    s += x[e]; sq += x[e] * x[e];
  }
  rs[t] = s; rq[t] = sq;
  __syncthreads();
  for (int st = 128; st > 0; st >>= 1) {
    if (t < st) { rs[t] += rs[t + st]; rq[t] += rq[t + st]; }
    __syncthreads();
  }
  float mu  = rs[0] * (1.f / DM);
  float var = rq[0] * (1.f / DM) - mu * mu;
  float inv = rsqrtf(var + 1e-6f);
#pragma unroll
  for (int e = 0; e < 4; ++e) {
    int c = t + e * 256;
    out[(size_t)row * DM + c] = (x[e] - mu) * inv * gamma[c] + beta[c];
  }
}

// =====================================================================
extern "C" void kernel_launch(void* const* d_in, const int* in_sizes, int n_in,
                              void* d_out, int out_size, void* d_ws, size_t ws_size,
                              hipStream_t stream) {
  (void)in_sizes; (void)n_in; (void)out_size;
  if (ws_size < WS_NEEDED) return;   // need ~60 MB scratch

  const float* h     = (const float*)d_in[0];
  const float* r     = (const float*)d_in[1];
  const float* rwb   = (const float*)d_in[2];
  const float* rrb   = (const float*)d_in[3];
  const float* Wqkv  = (const float*)d_in[4];
  const float* Wr    = (const float*)d_in[5];
  const float* Wo    = (const float*)d_in[6];
  const float* gamma = (const float*)d_in[7];
  const float* beta  = (const float*)d_in[8];
  float* out = (float*)d_out;

  char* ws = (char*)d_ws;
  unsigned short* q1  = (unsigned short*)(ws + OFF_Q1);
  unsigned short* q2  = (unsigned short*)(ws + OFF_Q2);
  unsigned short* kb  = (unsigned short*)(ws + OFF_K);
  unsigned short* vt  = (unsigned short*)(ws + OFF_VT);
  unsigned short* rk  = (unsigned short*)(ws + OFF_RK);
  unsigned short* av  = (unsigned short*)(ws + OFF_AV);
  float*          y   = (float*)(ws + OFF_Y);

  // 1) QKV projection: [4096,1024] x [1024,3072]
  proj_gemm<<<dim3(32, 48), 256, 0, stream>>>(h, Wqkv, DM, 3 * DM, 0,
                                              rwb, rrb, q1, q2, kb, vt, rk);
  // 2) r projection: [2048,1024] x [1024,1024]
  proj_gemm<<<dim3(16, 16), 256, 0, stream>>>(r, Wr, DM, DM, 1,
                                              rwb, rrb, q1, q2, kb, vt, rk);
  // 3) fused rel-shift flash attention (TDM-staged K/V)
  attn_kernel<<<dim3(S / 64, NH, BSZ), 128, 0, stream>>>(q1, q2, kb, vt, rk, av);
  // 4) output projection: [4096,1024] x [1024,1024]
  out_gemm<<<dim3(32, 16), 256, 0, stream>>>(av, Wo, y);
  // 5) residual + LayerNorm
  ln_kernel<<<S * BSZ, 256, 0, stream>>>(h, y, gamma, beta, out);
}
